// LSTMCell_64476049047569
// MI455X (gfx1250) — matrix operand order repaired and checked by reading, not compiled, not run on hardware
//
#include <hip/hip_runtime.h>
#include <cstdint>
#include <cstddef>

#define BATCH 4096
#define HID   2048
#define KDIM  4096   // INPUT_SIZE + HIDDEN_SIZE concatenated along K
#define G4H   8192   // 4 * HID

typedef __attribute__((ext_vector_type(16))) __bf16 v16bf;
typedef __attribute__((ext_vector_type(8)))  float  v8f;

union FragBF { v16bf v; uint4 q[2]; };

__device__ __forceinline__ unsigned short f2bf(float f) {
    union { float f; unsigned u; } cv; cv.f = f;
    unsigned u = cv.u;
    unsigned r = u + 0x7FFFu + ((u >> 16) & 1u);  // round-to-nearest-even
    return (unsigned short)(r >> 16);
}

__device__ __forceinline__ float sigmoid_f(float x) {
    return 1.0f / (1.0f + __expf(-x));
}

// out[r][0:C) = a[r][:] ; out[r][C:2C) = b[r][:]   (fp32 -> bf16)
__global__ void concat_cvt_bf16(const float* __restrict__ a,
                                const float* __restrict__ b,
                                unsigned short* __restrict__ out,
                                int rows, int C) {
    size_t t    = (size_t)blockIdx.x * blockDim.x + threadIdx.x;
    size_t base = t * 4;
    size_t total = (size_t)rows * 2 * (size_t)C;
    if (base >= total) return;
    int twoC  = 2 * C;
    size_t row = base / (size_t)twoC;
    int    c   = (int)(base % (size_t)twoC);
    const float* src = (c < C) ? (a + row * (size_t)C + c)
                               : (b + row * (size_t)C + (c - C));
    float4 f = *(const float4*)src;
    ushort4 o;
    o.x = f2bf(f.x); o.y = f2bf(f.y); o.z = f2bf(f.z); o.w = f2bf(f.w);
    *(ushort4*)(out + base) = o;
}

__global__ void bias_sum_kernel(const float* __restrict__ bx,
                                const float* __restrict__ bh,
                                float* __restrict__ bs) {
    int i = blockIdx.x * blockDim.x + threadIdx.x;
    if (i < G4H) bs[i] = bx[i] + bh[i];
}

// ---- GEMM tile geometry ----
#define BM  128
#define BHC 64
#define KC  32
#define LDA 40                 // halves, padded
#define LDW 40
#define ASZ (BM * LDA)         // halves per A buffer (5120)
#define WSZ (4 * BHC * LDW)    // halves per W buffer (10240)

// Issue one stage of async global->LDS copies (3 x B128 per thread).
__device__ __forceinline__ void issue_stage(unsigned ldsA, unsigned ldsW,
                                            const unsigned short* gA,
                                            const unsigned short* gW) {
    asm volatile("global_load_async_to_lds_b128 %0, %1, off"
                 :: "v"(ldsA), "v"(gA) : "memory");
    asm volatile("global_load_async_to_lds_b128 %0, %1, off"
                 :: "v"(ldsW), "v"(gW) : "memory");
    asm volatile("global_load_async_to_lds_b128 %0, %1, off"
                 :: "v"(ldsW + 16u), "v"(gW + 8) : "memory");
}

// 8 WMMAs on one staged K-slice.
__device__ __forceinline__ void mm_stage(const unsigned short* __restrict__ a0,
                                         const unsigned short* __restrict__ w0,
                                         v8f (&acc)[4][2],
                                         int wm, int wn, int l16, int lh) {
    FragBF fa[2], fb[4];
#pragma unroll
    for (int mt = 0; mt < 2; ++mt) {
        const unsigned short* p = a0 + (wm * 32 + mt * 16 + l16) * LDA;
        fa[mt].q[0] = *(const uint4*)(p + lh * 16);
        fa[mt].q[1] = *(const uint4*)(p + lh * 16 + 8);
    }
#pragma unroll
    for (int g = 0; g < 4; ++g) {
        const unsigned short* p = w0 + (g * BHC + wn * 16 + l16) * LDW + lh * 16;
        fb[g].q[0] = *(const uint4*)(p);
        fb[g].q[1] = *(const uint4*)(p + 8);
    }
#pragma unroll
    for (int g = 0; g < 4; ++g)
#pragma unroll
        for (int mt = 0; mt < 2; ++mt)
            acc[g][mt] = __builtin_amdgcn_wmma_f32_16x16x32_bf16(
                false, fa[mt].v, false, fb[g].v,
                (short)0, acc[g][mt], false, false);
}

// One block: 128 batch rows x 64 hidden cols x 4 gates, fused LSTM epilogue.
// 512 threads = 16 wave32s arranged 4 (M) x 4 (N). Double-buffered async-LDS.
__launch_bounds__(512)
__global__ void lstm_wmma_kernel(const unsigned short* __restrict__ Xcat,   // [4096][4096] bf16
                                 const unsigned short* __restrict__ Wcat,   // [8192][4096] bf16
                                 const float* __restrict__ biasSum,         // [8192]
                                 const float* __restrict__ cx,              // [4096][2048]
                                 float* __restrict__ out) {                 // hy | cy
    __shared__ __align__(16) unsigned short As[2 * ASZ];       // 20 KB
    __shared__ __align__(16) unsigned short Ws_[2 * WSZ];      // 40 KB

    const int tid  = threadIdx.x;
    const int lane = tid & 31;
    const int wid  = tid >> 5;
    const int wm   = wid & 3;        // 32-row slab within 128
    const int wn   = wid >> 2;       // 16-col slab within 64
    const int bm0  = blockIdx.y * BM;
    const int h0   = blockIdx.x * BHC;
    const int l16  = lane & 15, lh = lane >> 4;

    // staging assignments: A: 128 rows x (4 x 16B); W: 256 rows x (2 x 32B)
    const int arow = tid >> 2, aseg = tid & 3;
    const int wrow = tid >> 1, wseg = tid & 1;
    const int wg   = wrow >> 6, wnn = wrow & 63;
    const unsigned short* gA = Xcat + (size_t)(bm0 + arow) * KDIM + aseg * 8;
    const unsigned short* gW = Wcat + (size_t)(wg * HID + h0 + wnn) * KDIM + wseg * 16;

    // LDS byte addresses for this thread's async destinations (flat addr[31:0])
    const unsigned ldsA = (unsigned)(size_t)(As  + arow * LDA + aseg * 8);
    const unsigned ldsW = (unsigned)(size_t)(Ws_ + wrow * LDW + wseg * 16);
    constexpr unsigned ABUF = ASZ * 2u;   // byte stride between A buffers
    constexpr unsigned WBUF = WSZ * 2u;   // byte stride between W buffers

    v8f acc[4][2];
#pragma unroll
    for (int g = 0; g < 4; ++g)
#pragma unroll
        for (int mt = 0; mt < 2; ++mt)
#pragma unroll
            for (int i = 0; i < 8; ++i) acc[g][mt][i] = 0.0f;

    constexpr int NIT = KDIM / KC;    // 128 (even)

    issue_stage(ldsA, ldsW, gA, gW);  // prologue: fill buffer 0 (k0 = 0)

    for (int it = 0; it < NIT; it += 2) {
        // ---- even stage: compute buf0, prefetch buf1 ----
        issue_stage(ldsA + ABUF, ldsW + WBUF,
                    gA + (it + 1) * KC, gW + (it + 1) * KC);   // it+1 < NIT always
        asm volatile("s_wait_asynccnt 0x3" ::: "memory");       // buf0 of this wave done
        __syncthreads();                                        // buf0 of all waves done
        mm_stage(As, Ws_, acc, wm, wn, l16, lh);
        __syncthreads();                                        // all reads of buf0 done

        // ---- odd stage: compute buf1, prefetch buf0 ----
        if (it + 2 < NIT) {
            issue_stage(ldsA, ldsW, gA + (it + 2) * KC, gW + (it + 2) * KC);
            asm volatile("s_wait_asynccnt 0x3" ::: "memory");
        } else {
            asm volatile("s_wait_asynccnt 0x0" ::: "memory");
        }
        __syncthreads();
        mm_stage(As + ASZ, Ws_ + WSZ, acc, wm, wn, l16, lh);
        __syncthreads();
    }

    // Fused LSTM epilogue. C/D layout: lane -> N=l16, VGPR v -> M = v + 8*lh.
    const int col = h0 + wn * 16 + l16;
    const float bf_ = biasSum[0 * HID + col];
    const float bi_ = biasSum[1 * HID + col];
    const float bc_ = biasSum[2 * HID + col];
    const float bo_ = biasSum[3 * HID + col];
#pragma unroll
    for (int mt = 0; mt < 2; ++mt) {
        const int rb = bm0 + wm * 32 + mt * 16 + lh * 8;
#pragma unroll
        for (int v = 0; v < 8; ++v) {
            const int m = rb + v;
            float ft = sigmoid_f(acc[0][mt][v] + bf_);
            float it_ = sigmoid_f(acc[1][mt][v] + bi_);
            float ct = tanhf(acc[2][mt][v] + bc_);
            float ot = sigmoid_f(acc[3][mt][v] + bo_);
            float cxv = cx[(size_t)m * HID + col];
            float cyv = ft * cxv + it_ * ct;
            float hyv = ot * tanhf(cyv);
            out[(size_t)m * HID + col] = hyv;                         // hy
            out[(size_t)BATCH * HID + (size_t)m * HID + col] = cyv;   // cy
        }
    }
}

extern "C" void kernel_launch(void* const* d_in, const int* in_sizes, int n_in,
                              void* d_out, int out_size, void* d_ws, size_t ws_size,
                              hipStream_t stream) {
    (void)in_sizes; (void)n_in; (void)out_size; (void)ws_size;
    const float* x  = (const float*)d_in[0];
    const float* hx = (const float*)d_in[1];
    const float* cx = (const float*)d_in[2];
    const float* Wx = (const float*)d_in[3];
    const float* bx = (const float*)d_in[4];
    const float* Wh = (const float*)d_in[5];
    const float* bh = (const float*)d_in[6];
    float* out = (float*)d_out;

    // workspace layout: Xcat bf16 (32MB) | Wcat bf16 (64MB) | biasSum f32 (32KB)
    unsigned short* Xcat = (unsigned short*)d_ws;
    unsigned short* Wcat = Xcat + (size_t)BATCH * KDIM;
    float* biasSum = (float*)(Wcat + (size_t)G4H * KDIM);

    {
        size_t tot = (size_t)BATCH * KDIM / 4;   // 4 elems/thread
        concat_cvt_bf16<<<dim3((unsigned)((tot + 255) / 256)), 256, 0, stream>>>(
            x, hx, Xcat, BATCH, HID);
    }
    {
        size_t tot = (size_t)G4H * KDIM / 4;
        concat_cvt_bf16<<<dim3((unsigned)((tot + 255) / 256)), 256, 0, stream>>>(
            Wx, Wh, Wcat, G4H, HID);
    }
    bias_sum_kernel<<<dim3(G4H / 256), 256, 0, stream>>>(bx, bh, biasSum);

    lstm_wmma_kernel<<<dim3(HID / BHC, BATCH / BM), 512, 0, stream>>>(
        Xcat, Wcat, biasSum, cx, out);
}
